// SimpleSelfAttention_1030792151199
// MI455X (gfx1250) — matrix operand order repaired and checked by reading, not compiled, not run on hardware
//
#include <hip/hip_runtime.h>
#include <math.h>

// ---------------------------------------------------------------------------
// Types for CDNA5 WMMA (wave32): D(16x16 f32) = A(16x32 bf16) x B(32x16 bf16) + C
// ---------------------------------------------------------------------------
typedef __attribute__((ext_vector_type(16))) __bf16 v16bf;
typedef __attribute__((ext_vector_type(8)))  float  v8f;
typedef unsigned short u16;

union FragAB { v16bf v; uint4 q[2]; u16 u[16]; };
union FragC  { v8f  v; float f[8]; };

__device__ __forceinline__ u16 f2bf(float x) {
  union { float f; unsigned u; } cv; cv.f = x;
  unsigned r = cv.u + 0x7FFFu + ((cv.u >> 16) & 1u);  // round-to-nearest-even
  return (u16)(r >> 16);
}

__device__ __forceinline__ v8f wmma_bf16(const FragAB& a, const FragAB& b, v8f c) {
  return __builtin_amdgcn_wmma_f32_16x16x32_bf16(false, a.v, false, b.v,
                                                 (short)0, c, false, false);
}

// ---------------------------------------------------------------------------
// CDNA5 async global->LDS copy (GLOBAL_LOAD_ASYNC_TO_LDS_B128, ASYNCcnt).
// Per ISA 10.2: generic pointer to LDS has the LDS byte offset in addr[31:0],
// so a truncating cast gives the VDST (LDS address) operand.
// ---------------------------------------------------------------------------
#define USE_ASYNC_COPY 1

__device__ __forceinline__ void cp_async16(void* dst_lds, const void* src_glob) {
#if USE_ASYNC_COPY
  unsigned lds = (unsigned)(unsigned long long)dst_lds;
  unsigned long long g = (unsigned long long)src_glob;
  asm volatile("global_load_async_to_lds_b128 %0, %1, off"
               :: "v"(lds), "v"(g) : "memory");
#else
  *(uint4*)dst_lds = *(const uint4*)src_glob;
#endif
}

__device__ __forceinline__ void cp_async32(u16* dst_lds, const u16* src_glob) {
  cp_async16(dst_lds, src_glob);
  cp_async16(dst_lds + 8, src_glob + 8);
}

__device__ __forceinline__ void wait_async() {
#if USE_ASYNC_COPY
  asm volatile("s_wait_asynccnt 0x0" ::: "memory");
#endif
}

// ---------------------------------------------------------------------------
// Problem constants
// ---------------------------------------------------------------------------
#define HID   1024
#define HEADS 16
#define DHEAD 64
#define SEQ   2048
#define BATCH 2
#define ROWS  (BATCH * SEQ)        // 4096
#define NKQV  (3 * HID)            // 3072
#define BH    (BATCH * HEADS)      // 32

// ---------------------------------------------------------------------------
// f32 -> bf16 conversion
// ---------------------------------------------------------------------------
__global__ void convert_f32_bf16(const float* __restrict__ src,
                                 u16* __restrict__ dst, int n) {
  int i = blockIdx.x * blockDim.x + threadIdx.x;
  if (i < n) dst[i] = f2bf(src[i]);
}

// Transpose [K][N] f32 -> [N][K] bf16 via 32x32 LDS tiles (coalesced both sides)
__global__ __launch_bounds__(256)
void transpose_cvt(const float* __restrict__ src, u16* __restrict__ dst,
                   int K, int N) {
  __shared__ float tile[32][33];
  int n0 = blockIdx.x * 32, k0 = blockIdx.y * 32;
  int tx = threadIdx.x & 31, ty = threadIdx.x >> 5;   // ty = 0..7
#pragma unroll
  for (int j = 0; j < 32; j += 8)
    tile[ty + j][tx] = src[(size_t)(k0 + ty + j) * N + (n0 + tx)];
  __syncthreads();
#pragma unroll
  for (int j = 0; j < 32; j += 8)
    dst[(size_t)(n0 + ty + j) * K + (k0 + tx)] = f2bf(tile[tx][ty + j]);
}

// ---------------------------------------------------------------------------
// Tiled bf16 WMMA GEMM: C[M][N] = A[M][K] * BT[N][K]^T + bias
// 128(M) x 256(N) x 32(K) block tiles, double-buffered LDS staged with
// async global->LDS. 256 threads = 8 waves as 2(M)x4(N); each wave owns
// 64x64 = 4x4 WMMA accumulators (16 WMMA per K-step).
// MODE 0: kqv epilogue -> scatter to K[bh][s][d], Q[bh][s][d], Vt[bh][d][s] bf16
// MODE 1: ff  epilogue -> f32 output
// ---------------------------------------------------------------------------
#define LDSTRIDE 40   // 32 + 8 pad (80B rows: 16B aligned, bank-conflict friendly)

template <int MODE>
__global__ __launch_bounds__(256)
void gemm_bf16(const u16* __restrict__ A, const u16* __restrict__ BT,
               const float* __restrict__ bias, int M, int N, int K,
               u16* __restrict__ Kg, u16* __restrict__ Qg,
               u16* __restrict__ Vtg, float* __restrict__ Cout) {
  __shared__ u16 sA[2][128 * LDSTRIDE];
  __shared__ u16 sB[2][256 * LDSTRIDE];

  const int tid  = threadIdx.x;
  const int wave = tid >> 5, lane = tid & 31;
  const int half = lane >> 4, c16 = lane & 15;
  const int mW = (wave & 1) * 64;   // wave M offset in tile
  const int nW = (wave >> 1) * 64;  // wave N offset in tile
  const int rowBase = blockIdx.y * 128;
  const int colBase = blockIdx.x * 256;

  FragC acc[4][4];
#pragma unroll
  for (int mf = 0; mf < 4; ++mf)
#pragma unroll
    for (int nf = 0; nf < 4; ++nf)
#pragma unroll
      for (int i = 0; i < 8; ++i) acc[mf][nf].f[i] = 0.0f;

  // Async tile staging: A: thread t -> row t/2, 16-elem segment (t&1)*16.
  //                     B: thread t -> row t, all 32 elems.
  const int arow = tid >> 1, aseg = (tid & 1) * 16;
  auto issueTile = [&](int kt, int bsel) {
    const int k0 = kt * 32;
    const u16* ga = A + (size_t)(rowBase + arow) * K + k0 + aseg;
    cp_async32(&sA[bsel][arow * LDSTRIDE + aseg], ga);
    const u16* gb = BT + (size_t)(colBase + tid) * K + k0;
    cp_async32(&sB[bsel][tid * LDSTRIDE], gb);
    cp_async32(&sB[bsel][tid * LDSTRIDE + 16], gb + 16);
  };

  issueTile(0, 0);
  const int KT = K >> 5;
  for (int kt = 0; kt < KT; ++kt) {
    const int cur = kt & 1;
    wait_async();        // our async writes into sX[cur] are complete
    __syncthreads();     // everyone's writes visible; sX[cur^1] free again
    if (kt + 1 < KT) issueTile(kt + 1, cur ^ 1);

    FragAB af[4], bfr[4];
#pragma unroll
    for (int mf = 0; mf < 4; ++mf) {
      const u16* base = &sA[cur][(mW + mf * 16 + c16) * LDSTRIDE];
      af[mf].q[0] = *(const uint4*)(base + 8 * half);        // K = 8h .. 8h+7
      af[mf].q[1] = *(const uint4*)(base + 16 + 8 * half);   // K = 16+8h .. +7
    }
#pragma unroll
    for (int nf = 0; nf < 4; ++nf) {
      const u16* base = &sB[cur][(nW + nf * 16 + c16) * LDSTRIDE];
      bfr[nf].q[0] = *(const uint4*)(base + 8 * half);       // K = 8h .. 8h+7
      bfr[nf].q[1] = *(const uint4*)(base + 16 + 8 * half);  // K = 16+8h .. +7
    }
#pragma unroll
    for (int mf = 0; mf < 4; ++mf)
#pragma unroll
      for (int nf = 0; nf < 4; ++nf)
        acc[mf][nf].v = wmma_bf16(af[mf], bfr[nf], acc[mf][nf].v);
  }

  // epilogue
#pragma unroll
  for (int mf = 0; mf < 4; ++mf)
#pragma unroll
    for (int nf = 0; nf < 4; ++nf)
#pragma unroll
      for (int rr = 0; rr < 8; ++rr) {
        int gRow = rowBase + mW + mf * 16 + rr + 8 * half;
        int gCol = colBase + nW + nf * 16 + c16;
        float v = acc[mf][nf].f[rr] + bias[gCol];
        if (MODE == 0) {
          int mat = gCol >> 10, hid = gCol & 1023;
          int head = hid >> 6, d = hid & 63;
          int bidx = gRow >> 11, s = gRow & 2047;
          int bh = bidx * HEADS + head;
          if (mat == 0)
            Kg[((size_t)bh * SEQ + s) * DHEAD + d] = f2bf(v);
          else if (mat == 1)
            Qg[((size_t)bh * SEQ + s) * DHEAD + d] = f2bf(v);
          else
            Vtg[((size_t)bh * DHEAD + d) * SEQ + s] = f2bf(v);  // V transposed
        } else {
          Cout[(size_t)gRow * N + gCol] = v;
        }
      }
}

// ---------------------------------------------------------------------------
// Pass 1 of attention: per key-row softmax stats over the QUERY axis.
// Block = (bh, s-block of 128). scores[s][n] = scale * K_s . Q_n via WMMA.
// Online max / sumexp across 16 n-blocks; store m[s] and 1/l[s].
// 8 waves = 4(s) x 2(n), each wave 32(s) x 64(n) = 2x4 fragments.
// ---------------------------------------------------------------------------
__global__ __launch_bounds__(256)
void attn_stats(const u16* __restrict__ Kg, const u16* __restrict__ Qg,
                float* __restrict__ mG, float* __restrict__ rlG, float scale) {
  __shared__ u16 sK[128 * 64];   // [s][d]
  __shared__ u16 sQ[128 * 64];   // [n][d]  (Q row-major == B^T layout)
  __shared__ float mPart[2][128], lPart[2][128];

  const int bh = blockIdx.y, sBlk = blockIdx.x;
  const int tid = threadIdx.x;
  const int wave = tid >> 5, lane = tid & 31;
  const int half = lane >> 4, c16 = lane & 15;
  const int mW = (wave & 3) * 32;
  const int wn = wave >> 2;
  const int nW = wn * 64;

  const int lrow = tid >> 1, lseg = (tid & 1) * 32;  // loader: 32 bf16 / thread
  {  // K tile: 128 rows x 64 (async; first-iteration wait covers it)
    const u16* g = Kg + ((size_t)bh * SEQ + sBlk * 128 + lrow) * DHEAD + lseg;
    cp_async32(&sK[lrow * 64 + lseg], g);
    cp_async32(&sK[lrow * 64 + lseg + 16], g + 16);
  }

  float m_run[2][8], l_run[2][8];
#pragma unroll
  for (int sf = 0; sf < 2; ++sf)
#pragma unroll
    for (int rr = 0; rr < 8; ++rr) { m_run[sf][rr] = -3.0e38f; l_run[sf][rr] = 0.0f; }

  for (int nb = 0; nb < 16; ++nb) {
    __syncthreads();   // previous iteration done reading sQ
    {  // Q n-block
      const u16* g = Qg + ((size_t)bh * SEQ + nb * 128 + lrow) * DHEAD + lseg;
      cp_async32(&sQ[lrow * 64 + lseg], g);
      cp_async32(&sQ[lrow * 64 + lseg + 16], g + 16);
    }
    wait_async();
    __syncthreads();

    FragC acc[2][4];
#pragma unroll
    for (int sf = 0; sf < 2; ++sf)
#pragma unroll
      for (int nf = 0; nf < 4; ++nf)
#pragma unroll
        for (int i = 0; i < 8; ++i) acc[sf][nf].f[i] = 0.0f;

#pragma unroll
    for (int k0 = 0; k0 < 64; k0 += 32) {
      FragAB ak[2], bq[4];
#pragma unroll
      for (int sf = 0; sf < 2; ++sf) {
        const u16* base = &sK[(mW + sf * 16 + c16) * 64 + k0];
        ak[sf].q[0] = *(const uint4*)(base + 8 * half);
        ak[sf].q[1] = *(const uint4*)(base + 16 + 8 * half);
      }
#pragma unroll
      for (int nf = 0; nf < 4; ++nf) {
        const u16* base = &sQ[(nW + nf * 16 + c16) * 64 + k0];
        bq[nf].q[0] = *(const uint4*)(base + 8 * half);
        bq[nf].q[1] = *(const uint4*)(base + 16 + 8 * half);
      }
#pragma unroll
      for (int sf = 0; sf < 2; ++sf)
#pragma unroll
        for (int nf = 0; nf < 4; ++nf)
          acc[sf][nf].v = wmma_bf16(ak[sf], bq[nf], acc[sf][nf].v);
    }

    // online row stats (row = key position, reduce over 16 lanes = n columns)
#pragma unroll
    for (int sf = 0; sf < 2; ++sf)
#pragma unroll
      for (int rr = 0; rr < 8; ++rr) {
        float v0 = fmaxf(fmaxf(acc[sf][0].f[rr], acc[sf][1].f[rr]),
                         fmaxf(acc[sf][2].f[rr], acc[sf][3].f[rr])) * scale;
#pragma unroll
        for (int msk = 1; msk < 16; msk <<= 1)
          v0 = fmaxf(v0, __shfl_xor(v0, msk, 32));
        float mOld = m_run[sf][rr];
        float mNew = fmaxf(mOld, v0);
        float se = 0.0f;
#pragma unroll
        for (int nf = 0; nf < 4; ++nf)
          se += __expf(acc[sf][nf].f[rr] * scale - mNew);
#pragma unroll
        for (int msk = 1; msk < 16; msk <<= 1)
          se += __shfl_xor(se, msk, 32);
        l_run[sf][rr] = l_run[sf][rr] * __expf(mOld - mNew) + se;
        m_run[sf][rr] = mNew;
      }
  }

  if (c16 == 0) {
#pragma unroll
    for (int sf = 0; sf < 2; ++sf)
#pragma unroll
      for (int rr = 0; rr < 8; ++rr) {
        int row = mW + sf * 16 + rr + 8 * half;
        mPart[wn][row] = m_run[sf][rr];
        lPart[wn][row] = l_run[sf][rr];
      }
  }
  __syncthreads();
  if (tid < 128) {
    float m0 = mPart[0][tid], m1 = mPart[1][tid];
    float mM = fmaxf(m0, m1);
    float L = lPart[0][tid] * __expf(m0 - mM) + lPart[1][tid] * __expf(m1 - mM);
    size_t row = (size_t)bh * SEQ + sBlk * 128 + tid;
    mG[row]  = mM;
    rlG[row] = 1.0f / L;
  }
}

// ---------------------------------------------------------------------------
// Pass 2 of attention: O[n][d] = sum_s P[s][n] * V[s][d], P = exp(scale*sc - m_s)/l_s
// Block = (bh, n-block of 128). Loop 32 s-blocks of 64.
// Phase 1 (8 waves = 2(s) x 4(n), 32x32 regions): score tile + P^T -> LDS.
// Phase 2 (wave w owns n rows [16w,16w+16), all 64 d): O += P^T @ V  (V staged
// transposed [d][s] so B fragments are contiguous-K LDS reads).
// ---------------------------------------------------------------------------
__global__ __launch_bounds__(256)
void attn_out(const u16* __restrict__ Kg, const u16* __restrict__ Qg,
              const u16* __restrict__ Vtg, const float* __restrict__ mG,
              const float* __restrict__ rlG, u16* __restrict__ Og, float scale) {
  __shared__ u16 sQ[128 * 64];   // [n][d]
  __shared__ u16 sK[64 * 64];    // [s][d]
  __shared__ u16 sVt[64 * 64];   // [d][s]
  __shared__ u16 sPt[128 * 64];  // [n][s]  (P transposed, bf16)
  __shared__ float sM[64], sRL[64];

  const int bh = blockIdx.y, nBlk = blockIdx.x;
  const int tid = threadIdx.x;
  const int wave = tid >> 5, lane = tid & 31;
  const int half = lane >> 4, c16 = lane & 15;
  const int p1s = (wave & 1) * 32;
  const int p1n = (wave >> 1) * 32;

  FragC oacc[4];
#pragma unroll
  for (int dt = 0; dt < 4; ++dt)
#pragma unroll
    for (int i = 0; i < 8; ++i) oacc[dt].f[i] = 0.0f;

  {  // Q n-block, loaded once (async; first in-loop wait covers it)
    int rr = tid >> 1, h = (tid & 1) * 32;
    const u16* g = Qg + ((size_t)bh * SEQ + nBlk * 128 + rr) * DHEAD + h;
    cp_async32(&sQ[rr * 64 + h], g);
    cp_async32(&sQ[rr * 64 + h + 16], g + 16);
  }

  for (int sb = 0; sb < 32; ++sb) {
    __syncthreads();   // previous iteration done with sK/sVt/sPt/sM
    {  // K s-block [64][64] and Vt s-block [64 d][64 s]: 16 bf16 per thread each
      int rr = tid >> 2, h = (tid & 3) * 16;
      cp_async32(&sK[rr * 64 + h],
                 Kg + ((size_t)bh * SEQ + sb * 64 + rr) * DHEAD + h);
      cp_async32(&sVt[rr * 64 + h],
                 Vtg + ((size_t)bh * DHEAD + rr) * SEQ + sb * 64 + h);
    }
    if (tid < 64) {
      sM[tid]  = mG[(size_t)bh * SEQ + sb * 64 + tid];
      sRL[tid] = rlG[(size_t)bh * SEQ + sb * 64 + tid];
    }
    wait_async();
    __syncthreads();

    // ---- phase 1: score tile [64 s][128 n], this wave's 32x32 region
    FragC acc[2][2];
#pragma unroll
    for (int sf = 0; sf < 2; ++sf)
#pragma unroll
      for (int nf = 0; nf < 2; ++nf)
#pragma unroll
        for (int i = 0; i < 8; ++i) acc[sf][nf].f[i] = 0.0f;

#pragma unroll
    for (int k0 = 0; k0 < 64; k0 += 32) {
      FragAB ak[2], bq[2];
#pragma unroll
      for (int sf = 0; sf < 2; ++sf) {
        const u16* base = &sK[(p1s + sf * 16 + c16) * 64 + k0];
        ak[sf].q[0] = *(const uint4*)(base + 8 * half);
        ak[sf].q[1] = *(const uint4*)(base + 16 + 8 * half);
      }
#pragma unroll
      for (int nf = 0; nf < 2; ++nf) {
        const u16* base = &sQ[(p1n + nf * 16 + c16) * 64 + k0];
        bq[nf].q[0] = *(const uint4*)(base + 8 * half);
        bq[nf].q[1] = *(const uint4*)(base + 16 + 8 * half);
      }
#pragma unroll
      for (int sf = 0; sf < 2; ++sf)
#pragma unroll
        for (int nf = 0; nf < 2; ++nf)
          acc[sf][nf].v = wmma_bf16(ak[sf], bq[nf], acc[sf][nf].v);
    }

    // P = exp(scale*score - m_s) * (1/l_s), stored transposed [n][s] as bf16
#pragma unroll
    for (int sf = 0; sf < 2; ++sf)
#pragma unroll
      for (int nf = 0; nf < 2; ++nf)
#pragma unroll
        for (int rr = 0; rr < 8; ++rr) {
          int sl = p1s + sf * 16 + rr + 8 * half;
          int nl = p1n + nf * 16 + c16;
          float p = __expf(acc[sf][nf].f[rr] * scale - sM[sl]) * sRL[sl];
          sPt[nl * 64 + sl] = f2bf(p);
        }
    __syncthreads();

    // ---- phase 2: O[16 n rows of this wave][64 d] += P^T @ V
#pragma unroll
    for (int k0 = 0; k0 < 64; k0 += 32) {
      FragAB ap;
      const u16* abase = &sPt[(wave * 16 + c16) * 64 + k0];
      ap.q[0] = *(const uint4*)(abase + 8 * half);
      ap.q[1] = *(const uint4*)(abase + 16 + 8 * half);
#pragma unroll
      for (int dt = 0; dt < 4; ++dt) {
        FragAB bv;
        const u16* bbase = &sVt[(dt * 16 + c16) * 64 + k0];
        bv.q[0] = *(const uint4*)(bbase + 8 * half);
        bv.q[1] = *(const uint4*)(bbase + 16 + 8 * half);
        oacc[dt].v = wmma_bf16(ap, bv, oacc[dt].v);
      }
    }
  }

  // epilogue: Og[b][n][head*64+d] bf16 (feeds the ff GEMM)
  const int bidx = bh >> 4, head = bh & 15;
#pragma unroll
  for (int dt = 0; dt < 4; ++dt)
#pragma unroll
    for (int rr = 0; rr < 8; ++rr) {
      int n = nBlk * 128 + wave * 16 + rr + 8 * half;
      int d = dt * 16 + c16;
      Og[((size_t)(bidx * SEQ + n)) * HID + head * DHEAD + d] = f2bf(oacc[dt].f[rr]);
    }
}

// ---------------------------------------------------------------------------
// Launch. Workspace layout (~51 MB): Xb, WkqvT, WffT, K, Q, Vt, Og, m, 1/l
// ---------------------------------------------------------------------------
extern "C" void kernel_launch(void* const* d_in, const int* in_sizes, int n_in,
                              void* d_out, int out_size, void* d_ws, size_t ws_size,
                              hipStream_t stream) {
  const float* X    = (const float*)d_in[0];  // [2,2048,1024]
  const float* Wkqv = (const float*)d_in[1];  // [1024,3072]
  const float* bkqv = (const float*)d_in[2];  // [3072]
  const float* Wff  = (const float*)d_in[3];  // [1024,1024]
  const float* bff  = (const float*)d_in[4];  // [1024]
  float* out = (float*)d_out;

  char* ws = (char*)d_ws;
  size_t off = 0;
  auto carve = [&](size_t bytes) -> void* {
    void* p = ws + off;
    off = (off + bytes + 255) & ~(size_t)255;
    return p;
  };
  u16*   Xb  = (u16*)carve((size_t)ROWS * HID * 2);
  u16*   WkT = (u16*)carve((size_t)NKQV * HID * 2);
  u16*   WfT = (u16*)carve((size_t)HID * HID * 2);
  u16*   Kg  = (u16*)carve((size_t)BH * SEQ * DHEAD * 2);
  u16*   Qg  = (u16*)carve((size_t)BH * SEQ * DHEAD * 2);
  u16*   Vtg = (u16*)carve((size_t)BH * SEQ * DHEAD * 2);
  u16*   Og  = (u16*)carve((size_t)ROWS * HID * 2);
  float* mG  = (float*)carve((size_t)BH * SEQ * 4);
  float* rlG = (float*)carve((size_t)BH * SEQ * 4);

  const float scale = 0.25f;  // 1024^(-0.2) = 2^(-2) exactly

  convert_f32_bf16<<<(ROWS * HID) / 256, 256, 0, stream>>>(X, Xb, ROWS * HID);
  transpose_cvt<<<dim3(NKQV / 32, HID / 32), 256, 0, stream>>>(Wkqv, WkT, HID, NKQV);
  transpose_cvt<<<dim3(HID / 32, HID / 32), 256, 0, stream>>>(Wff, WfT, HID, HID);

  gemm_bf16<0><<<dim3(NKQV / 256, ROWS / 128), 256, 0, stream>>>(
      Xb, WkT, bkqv, ROWS, NKQV, HID, Kg, Qg, Vtg, nullptr);

  attn_stats<<<dim3(SEQ / 128, BH), 256, 0, stream>>>(Kg, Qg, mG, rlG, scale);
  attn_out<<<dim3(SEQ / 128, BH), 256, 0, stream>>>(Kg, Qg, Vtg, mG, rlG, Og, scale);

  gemm_bf16<1><<<dim3(HID / 256, ROWS / 128), 256, 0, stream>>>(
      Og, WfT, bff, ROWS, HID, HID, nullptr, nullptr, nullptr, out);

  (void)in_sizes; (void)n_in; (void)out_size; (void)ws_size;
}
